// Layer2D_55525337203215
// MI455X (gfx1250) — compile-verified, matrix operand order checked by reading, not run on hardware
//
#include <hip/hip_runtime.h>
#include <cstdint>
#include <cstddef>

typedef __attribute__((ext_vector_type(16))) _Float16 v16h;
typedef __attribute__((ext_vector_type(8)))  float    v8f;

#define TPB 256
#define DEV __device__ __forceinline__

// Padded leading dimensions (halves). Row pitch == 4 mod 16 dwords ->
// conflict-free ds_load_b128 fragment loads across 16 lanes; 16B aligned.
#define LD128 136   // 272B  = 68 dw
#define LD96  104   // 208B  = 52 dw
#define LD32  40    // 80B   = 20 dw
#define LD512 520   // 1040B = 260 dw

DEV float gelu_tanh(float x) {
  const float c0 = 0.7978845608028654f;
  const float c1 = 0.044715f;
  return 0.5f * x * (1.f + tanhf(c0 * (x + c1 * x * x * x)));
}

DEV float wave_sum(float v) {
#pragma unroll
  for (int off = 16; off > 0; off >>= 1) v += __shfl_xor(v, off, 32);
  return v;
}
DEV float wave_max(float v) {
#pragma unroll
  for (int off = 16; off > 0; off >>= 1) v = fmaxf(v, __shfl_xor(v, off, 32));
  return v;
}

// Load a 16x32 f16 fragment (A layout; also used for B via transposed weights,
// i.e. Bt[n][k] row-major). CDNA5 wave32 layout:
// lanes 0-15: rows r0..r0+15, K pairs {0..7,16..23}; lanes 16-31: same rows, +8.
DEV v16h load_frag(const _Float16* M, int ld, int r0, int k0, int lane) {
  v16h f;
  const _Float16* row = M + (size_t)(r0 + (lane & 15)) * ld + k0 + ((lane & 16) ? 8 : 0);
#pragma unroll
  for (int j = 0; j < 8; ++j) {
    const int k = (j < 4) ? (2 * j) : (16 + 2 * (j - 4));
    f[2 * j]     = row[k];
    f[2 * j + 1] = row[k + 1];
  }
  return f;
}

// Workgroup-wide GEMM: C(Mt*16 x Nt*16) = epilogue(alpha * A @ Bt^T).
// A row-major (M x K); Bt row-major (N x K). Tiles round-robined over waves.
// Epilogue fully compile-time so it vectorizes (no per-element branches).
template <int ACT, bool ROWSB, bool COLB, bool RES, bool ST32, bool ST16,
          bool STORET, bool ACCUM>
DEV void wave_gemm(const _Float16* A, int lda,
                   const _Float16* Bt, int ldb,
                   int Mt, int Nt, int K, float alpha,
                   const float* rowScale, const float* rowBias,
                   const float* colBias,
                   const float* resid, int ldres,
                   float* Cf32, int ldc32,
                   _Float16* Cf16, int ldc16)
{
  const int tid  = threadIdx.x;
  const int wave = tid >> 5;
  const int lane = tid & 31;
  const int nwave = blockDim.x >> 5;
  const int hi = (lane >> 4) << 3;
  const int cN = lane & 15;
  for (int t = wave; t < Mt * Nt; t += nwave) {
    const int m0 = (t / Nt) * 16;
    const int n0 = (t % Nt) * 16;
    const int rbase = m0 + hi;
    const int cc = n0 + cN;
    v8f acc;
#pragma unroll
    for (int r = 0; r < 8; ++r)
      acc[r] = ACCUM ? Cf32[(size_t)(rbase + r) * ldc32 + cc] : 0.f;
    for (int k0 = 0; k0 < K; k0 += 32) {
      v16h a = load_frag(A,  lda, m0, k0, lane);
      v16h b = load_frag(Bt, ldb, n0, k0, lane);
      acc = __builtin_amdgcn_wmma_f32_16x16x32_f16(false, a, false, b,
                                                   (short)0, acc, false, false);
    }
    float rsv[8], rbv[8], rez[8];
    if (ROWSB) {
#pragma unroll
      for (int r = 0; r < 8; ++r) {
        rsv[r] = rowScale[rbase + r];
        rbv[r] = rowBias[rbase + r];
      }
    }
    const float cb = COLB ? colBias[cc] : 0.f;
    if (RES) {
#pragma unroll
      for (int r = 0; r < 8; ++r) rez[r] = resid[(size_t)(rbase + r) * ldres + cc];
    }
#pragma unroll
    for (int r = 0; r < 8; ++r) {
      float v = acc[r] * alpha;
      if (ROWSB) v = v * rsv[r] + rbv[r];
      if (COLB)  v += cb;
      if (ACT == 1) v = fmaxf(v, 0.f);
      else if (ACT == 2) v = gelu_tanh(v);
      if (RES) v += rez[r];
      if (ST32) {
        if (STORET) Cf32[(size_t)cc * ldc32 + rbase + r] = v;
        else        Cf32[(size_t)(rbase + r) * ldc32 + cc] = v;
      }
      if (ST16) {
        if (STORET) Cf16[(size_t)cc * ldc16 + rbase + r] = (_Float16)v;
        else        Cf16[(size_t)(rbase + r) * ldc16 + cc] = (_Float16)v;
      }
    }
  }
}

// LayerNorm over 80x128 LDS tile (x pitch 128 f32), one wave per row.
// WRITE_F32: write normalized value back to x (residual = LN output).
// ADD_B2: x += b2 (FF output-bias pre-accumulation), x otherwise untouched.
template <bool WRITE_F32, bool ADD_B2>
DEV void layernorm80(float* x, _Float16* xn, int ldxn,
                     const float* g, const float* be, const float* b2)
{
  const int wave = threadIdx.x >> 5;
  const int lane = threadIdx.x & 31;
  for (int r = wave; r < 80; r += 8) {
    float* xr = x + r * 128;
    _Float16* nr = xn + r * ldxn;
    if (r < 68) {
      float s = 0.f;
#pragma unroll
      for (int i = 0; i < 4; ++i) s += xr[lane + 32 * i];
      const float m = wave_sum(s) * (1.f / 128.f);
      float vv = 0.f;
#pragma unroll
      for (int i = 0; i < 4; ++i) {
        const float d = xr[lane + 32 * i] - m;
        vv += d * d;
      }
      const float rs = rsqrtf(wave_sum(vv) * (1.f / 128.f) + 1e-5f);
#pragma unroll
      for (int i = 0; i < 4; ++i) {
        const int c = lane + 32 * i;
        const float t = (xr[c] - m) * rs * g[c] + be[c];
        nr[c] = (_Float16)t;
        if (WRITE_F32) xr[c] = t;
        if (ADD_B2) xr[c] += b2[c];
      }
    } else {
#pragma unroll
      for (int i = 0; i < 4; ++i) {
        const int c = lane + 32 * i;
        nr[c] = (_Float16)0.f;
        if (WRITE_F32) xr[c] = 0.f;
      }
    }
  }
}

// ---------------------------------------------------------------------------
// Weight convert / transpose to f16 (B-operands stored transposed: Bt[n][k]).
__global__ void cvt_f16(const float* __restrict__ src, _Float16* __restrict__ dst,
                        int rows, int cols, int doT) {
  const int idx = blockIdx.x * blockDim.x + threadIdx.x;
  if (idx >= rows * cols) return;
  const float v = src[idx];
  if (doT) dst[(size_t)(idx % cols) * rows + (idx / cols)] = (_Float16)v;
  else     dst[idx] = (_Float16)v;
}

// ---------------------------------------------------------------------------
// Fused conv: pw1(512x128)+scale/bias+relu -> dw3x3+scale/bias+relu ->
// pw2(128x512)+scale/bias + x.  One block per 8x8 tile (== attention window).
__global__ void __launch_bounds__(TPB) conv_fused(
    const float* __restrict__ x,
    const _Float16* __restrict__ pw1h, const float* __restrict__ pw1_g, const float* __restrict__ pw1_b,
    const float* __restrict__ dw_w,   const float* __restrict__ dw_g,  const float* __restrict__ dw_b,
    const _Float16* __restrict__ pw2h, const float* __restrict__ pw2_g, const float* __restrict__ pw2_b,
    float* __restrict__ t_blk)
{
  extern __shared__ char smem[];
  _Float16* xh  = (_Float16*)smem;                 // [112][LD128] 30464B
  _Float16* hb  = (_Float16*)(smem + 30464);       // [512][112]   114688B
  _Float16* hdT = (_Float16*)(smem + 145152);      // [64][LD512]  66560B
  float*    dwc = (float*)(smem + 211712);         // [512*9]      18432B
  float*    oc  = (float*)(smem + 230144);         // [128][64]    32768B
  float*    dwg = (float*)(smem + 262912);         // 2048B
  float*    dwb = (float*)(smem + 264960);         // 2048B  (total 267008)

  const int tid = threadIdx.x;
  const int bid = blockIdx.x;
  const int b   = bid >> 10;
  const int win = bid & 1023;
  const int h0  = (win >> 5) << 3;
  const int w0  = (win & 31) << 3;

  // 10x10 halo, zero-padded; stored pixel-major (== Bt for GEMM1)
  for (int idx = tid; idx < 128 * 128; idx += TPB) {
    const int c = idx >> 7, q = idx & 127;
    if (q < 100) {
      const int hy = h0 + q / 10 - 1, wx = w0 + q % 10 - 1;
      float v = 0.f;
      if (hy >= 0 && hy < 256 && wx >= 0 && wx < 256)
        v = x[(((size_t)b * 128 + c) * 256 + hy) * 256 + wx];
      xh[q * LD128 + c] = (_Float16)v;
    }
  }
  for (int idx = tid; idx < 12 * 128; idx += TPB)
    xh[(100 + (idx >> 7)) * LD128 + (idx & 127)] = (_Float16)0.f;
  for (int idx = tid; idx < 512 * 9; idx += TPB) dwc[idx] = dw_w[idx];
  for (int idx = tid; idx < 512; idx += TPB) { dwg[idx] = dw_g[idx]; dwb[idx] = dw_b[idx]; }
  __syncthreads();

  wave_gemm<1, true, false, false, false, true, false, false>(
      pw1h, 128, xh, LD128, 32, 7, 128, 1.f,
      pw1_g, pw1_b, nullptr, nullptr, 0, nullptr, 0, hb, 112);
  __syncthreads();

  // depthwise 3x3 (cross-correlation) + scale/bias + relu; pixel-major out
  for (int idx = tid; idx < 512 * 64; idx += TPB) {
    const int o = idx >> 6, p = idx & 63;
    const int pi = p >> 3, pj = p & 7;
    float acc = 0.f;
#pragma unroll
    for (int di = 0; di < 3; ++di)
#pragma unroll
      for (int dj = 0; dj < 3; ++dj)
        acc += (float)hb[o * 112 + (pi + di) * 10 + (pj + dj)] * dwc[o * 9 + di * 3 + dj];
    const float v = fmaxf(acc * dwg[o] + dwb[o], 0.f);
    hdT[p * LD512 + o] = (_Float16)v;
  }
  __syncthreads();

  wave_gemm<0, true, false, false, true, false, false, false>(
      pw2h, 512, hdT, LD512, 8, 4, 512, 1.f,
      pw2_g, pw2_b, nullptr, nullptr, 0, oc, 64, nullptr, 0);
  __syncthreads();

  // + residual x, scatter into block-window token layout [seq][tok][128]
  for (int idx = tid; idx < 128 * 64; idx += TPB) {
    const int c = idx >> 6, p = idx & 63;
    const int hy = h0 + (p >> 3), wx = w0 + (p & 7);
    const float v = oc[idx] + x[(((size_t)b * 128 + c) * 256 + hy) * 256 + wx];
    t_blk[((size_t)bid * 64 + p) * 128 + c] = v;
  }
}

// ---------------------------------------------------------------------------
// Whole transformer stage for one sequence (T=68 -> padded 80).
// MODE 0 = block windows, MODE 1 = grid windows.
template <int MODE>
__global__ void __launch_bounds__(TPB) xformer(
    const float* __restrict__ tsrc,   // MODE0: t_blk ; MODE1: img (b,h,w,c)
    const float* __restrict__ regsrc, // MODE0: reg(4,128) ; MODE1: ravg(b,4,128)
    const _Float16* __restrict__ wqT, const _Float16* __restrict__ wkT,
    const _Float16* __restrict__ wvT, const _Float16* __restrict__ woT,
    const float* __restrict__ ln_g, const float* __restrict__ ln_b,
    const float* __restrict__ ff_lg, const float* __restrict__ ff_lb,
    const _Float16* __restrict__ w1T, const float* __restrict__ ff_b1,
    const _Float16* __restrict__ w2T, const float* __restrict__ ff_b2,
    float* __restrict__ outp, float* __restrict__ regout)
{
  extern __shared__ char smem[];
  float*    xin = (float*)smem;                  // [80][128] f32 (xl -> x1 -> y)
  _Float16* xn  = (_Float16*)(smem + 40960);     // [80][LD128] 21760B
  _Float16* qb  = (_Float16*)(smem + 62720);     // [80][LD128] 21760B
  _Float16* kf  = (_Float16*)(smem + 84480);     // [80][LD32]  6400B
  _Float16* vT  = (_Float16*)(smem + 90880);     // [32][LD96]  6656B
  float*    sb  = (float*)(smem + 97536);        // [80][80]    25600B
  _Float16* ab  = (_Float16*)(smem + 123136);    // [80][LD96]  16640B
  _Float16* ob  = (_Float16*)(smem + 139776);    // [80][LD128] 21760B
  _Float16* hbf = (_Float16*)(smem + 161536);    // [80][LD128] 21760B (total 183296)

  const int tid = threadIdx.x;
  const int wave = tid >> 5;
  const int lane = tid & 31;
  const int s   = blockIdx.x;
  const int b   = s >> 10;
  const int win = s & 1023;
  const int wa  = win >> 5, wb = win & 31;

  // gather tokens (regs 0..3 + 64 pixels), zero pad rows
  for (int idx = tid; idx < 80 * 128; idx += TPB) {
    const int r = idx >> 7, c = idx & 127;
    float v = 0.f;
    if (r < 68) {
      if (r < 4) {
        v = (MODE == 0) ? regsrc[r * 128 + c]
                        : regsrc[((size_t)b * 4 + r) * 128 + c];
      } else {
        const int t = r - 4;
        if (MODE == 0) {
          v = tsrc[((size_t)s * 64 + t) * 128 + c];
        } else {
          const int ph = (t >> 3) * 32 + wa, pw = (t & 7) * 32 + wb;
          v = tsrc[(((size_t)b * 256 + ph) * 256 + pw) * 128 + c];
        }
      }
    }
    xin[idx] = v;
  }
  for (int idx = tid; idx < 32 * LD96; idx += TPB) vT[idx] = (_Float16)0.f;
  __syncthreads();

  layernorm80<true, false>(xin, xn, LD128, ln_g, ln_b, nullptr);
  __syncthreads();

  // Q (pre-scaled by dh^-0.5), K, V (V stored transposed -> Bt for A@V)
  wave_gemm<0, false, false, false, false, true, false, false>(
      xn, LD128, wqT, 128, 5, 8, 128, 0.1767766952966369f,
      nullptr, nullptr, nullptr, nullptr, 0, nullptr, 0, qb, LD128);
  wave_gemm<0, false, false, false, false, true, false, false>(
      xn, LD128, wkT, 128, 5, 2, 128, 1.f,
      nullptr, nullptr, nullptr, nullptr, 0, nullptr, 0, kf, LD32);
  wave_gemm<0, false, false, false, false, true, true, false>(
      xn, LD128, wvT, 128, 5, 2, 128, 1.f,
      nullptr, nullptr, nullptr, nullptr, 0, nullptr, 0, vT, LD96);
  __syncthreads();

  for (int hh = 0; hh < 4; ++hh) {
    // S = q_h @ K^T  (kf is (token x dh) == Bt)
    wave_gemm<0, false, false, false, true, false, false, false>(
        qb + hh * 32, LD128, kf, LD32, 5, 5, 32, 1.f,
        nullptr, nullptr, nullptr, nullptr, 0, sb, 80, nullptr, 0);
    __syncthreads();
    // softmax: one wave per row, lane reductions
    for (int r = wave; r < 80; r += 8) {
      if (r < 68) {
        float mx = -1e30f;
        for (int c = lane; c < 68; c += 32) mx = fmaxf(mx, sb[r * 80 + c]);
        mx = wave_max(mx);
        float sum = 0.f;
        for (int c = lane; c < 68; c += 32) {
          const float e = __expf(sb[r * 80 + c] - mx);
          sb[r * 80 + c] = e;
          sum += e;
        }
        const float inv = 1.f / wave_sum(sum);
        for (int c = lane; c < LD96; c += 32)
          ab[r * LD96 + c] = (c < 68) ? (_Float16)(sb[r * 80 + c] * inv)
                                      : (_Float16)0.f;
      } else {
        for (int c = lane; c < LD96; c += 32) ab[r * LD96 + c] = (_Float16)0.f;
      }
    }
    __syncthreads();
    // O_h = A @ V  (K padded to 96 with zeros in ab cols / vT cols)
    wave_gemm<0, false, false, false, false, true, false, false>(
        ab, LD96, vT, LD96, 5, 2, 96, 1.f,
        nullptr, nullptr, nullptr, nullptr, 0, nullptr, 0, ob + hh * 32, LD128);
    __syncthreads();
  }

  // out-proj + residual(xl) -> x1 (in-place in xin)
  wave_gemm<0, false, false, true, true, false, false, false>(
      ob, LD128, woT, 128, 5, 8, 128, 1.f,
      nullptr, nullptr, nullptr, xin, 128, xin, 128, nullptr, 0);
  __syncthreads();

  // LN2 -> xn ; y init: xin = x1 + b2
  layernorm80<false, true>(xin, xn, LD128, ff_lg, ff_lb, ff_b2);
  __syncthreads();

  // FF: y += gelu(xn @ w1_chunk + b1_chunk) @ w2_chunk, 4 chunks of 128
  for (int ch = 0; ch < 4; ++ch) {
    wave_gemm<2, false, true, false, false, true, false, false>(
        xn, LD128, w1T + (size_t)ch * 128 * 128, 128, 5, 8, 128, 1.f,
        nullptr, nullptr, ff_b1 + ch * 128, nullptr, 0, nullptr, 0, hbf, LD128);
    __syncthreads();
    wave_gemm<0, false, false, false, true, false, false, true>(
        hbf, LD128, w2T + ch * 128, 512, 5, 8, 128, 1.f,
        nullptr, nullptr, nullptr, nullptr, 0, xin, 128, nullptr, 0);
    __syncthreads();
  }

  // scatter result
  if (MODE == 0) {
    for (int idx = tid; idx < 68 * 128; idx += TPB) {
      const int r = idx >> 7, c = idx & 127;
      const float v = xin[idx];
      if (r < 4) {
        regout[((size_t)s * 4 + r) * 128 + c] = v;
      } else {
        const int t = r - 4;
        const int ph = wa * 8 + (t >> 3), pw = wb * 8 + (t & 7);
        outp[(((size_t)b * 256 + ph) * 256 + pw) * 128 + c] = v;
      }
    }
  } else {
    for (int idx = tid; idx < 64 * 128; idx += TPB) {
      const int t = idx >> 7, c = idx & 127;
      const int ph = wa * 8 + (t >> 3), pw = wb * 8 + (t & 7);
      outp[(((size_t)b * 256 + ph) * 256 + pw) * 128 + c] = xin[(t + 4) * 128 + c];
    }
  }
}

// ---------------------------------------------------------------------------
__global__ void reg_avg(const float* __restrict__ rtmp, float* __restrict__ ravg) {
  const int idx = blockIdx.x * blockDim.x + threadIdx.x;
  if (idx >= 4 * 4 * 128) return;
  const int b = idx >> 9, rc = idx & 511;
  float acc = 0.f;
  for (int w = 0; w < 1024; ++w)
    acc += rtmp[(size_t)(b * 1024 + w) * 512 + rc];
  ravg[idx] = acc * (1.f / 1024.f);
}

// NHWC -> NCHW through an LDS tile (pitch 129 floats to avoid bank conflicts).
__global__ void __launch_bounds__(TPB) to_nchw(const float* __restrict__ img,
                                               float* __restrict__ out) {
  __shared__ float tile[32 * 129];
  const int bid = blockIdx.x;       // b*2048 + h*8 + wt
  const int wt  = bid & 7;
  const int h   = (bid >> 3) & 255;
  const int b   = bid >> 11;
  const int w0  = wt * 32;
  for (int idx = threadIdx.x; idx < 32 * 128; idx += TPB) {
    const int t = idx >> 7, c = idx & 127;
    tile[t * 129 + c] = img[(((size_t)b * 256 + h) * 256 + w0 + t) * 128 + c];
  }
  __syncthreads();
  for (int idx = threadIdx.x; idx < 32 * 128; idx += TPB) {
    const int c = idx >> 5, t = idx & 31;
    out[(((size_t)b * 128 + c) * 256 + h) * 256 + w0 + t] = tile[t * 129 + c];
  }
}

// ---------------------------------------------------------------------------
extern "C" void kernel_launch(void* const* d_in, const int* in_sizes, int n_in,
                              void* d_out, int out_size, void* d_ws, size_t ws_size,
                              hipStream_t stream) {
  (void)in_sizes; (void)n_in; (void)out_size; (void)ws_size;
  const float* x      = (const float*)d_in[0];
  const float* reg    = (const float*)d_in[1];
  const float* pw1_w  = (const float*)d_in[2];
  const float* pw1_g  = (const float*)d_in[3];
  const float* pw1_b  = (const float*)d_in[4];
  const float* dw_w   = (const float*)d_in[5];
  const float* dw_g   = (const float*)d_in[6];
  const float* dw_b   = (const float*)d_in[7];
  const float* pw2_w  = (const float*)d_in[8];
  const float* pw2_g  = (const float*)d_in[9];
  const float* pw2_b  = (const float*)d_in[10];
  const float* bln_g  = (const float*)d_in[11];
  const float* bln_b  = (const float*)d_in[12];
  const float* b_wq   = (const float*)d_in[13];
  const float* b_wk   = (const float*)d_in[14];
  const float* b_wv   = (const float*)d_in[15];
  const float* b_wo   = (const float*)d_in[16];
  const float* bff_lg = (const float*)d_in[17];
  const float* bff_lb = (const float*)d_in[18];
  const float* bff_w1 = (const float*)d_in[19];
  const float* bff_b1 = (const float*)d_in[20];
  const float* bff_w2 = (const float*)d_in[21];
  const float* bff_b2 = (const float*)d_in[22];
  const float* gln_g  = (const float*)d_in[23];
  const float* gln_b  = (const float*)d_in[24];
  const float* g_wq   = (const float*)d_in[25];
  const float* g_wk   = (const float*)d_in[26];
  const float* g_wv   = (const float*)d_in[27];
  const float* g_wo   = (const float*)d_in[28];
  const float* gff_lg = (const float*)d_in[29];
  const float* gff_lb = (const float*)d_in[30];
  const float* gff_w1 = (const float*)d_in[31];
  const float* gff_b1 = (const float*)d_in[32];
  const float* gff_w2 = (const float*)d_in[33];
  const float* gff_b2 = (const float*)d_in[34];

  char* ws = (char*)d_ws;
  size_t off = 0;
  auto carve = [&](size_t bytes) -> char* {
    char* p = ws + off;
    off = (off + bytes + 255) & ~(size_t)255;
    return p;
  };

  _Float16* pw1h = (_Float16*)carve((size_t)512 * 128 * 2);
  _Float16* pw2h = (_Float16*)carve((size_t)128 * 512 * 2);
  _Float16* bwqT = (_Float16*)carve((size_t)128 * 128 * 2);
  _Float16* bwkT = (_Float16*)carve((size_t)128 * 32 * 2);
  _Float16* bwvT = (_Float16*)carve((size_t)128 * 32 * 2);
  _Float16* bwoT = (_Float16*)carve((size_t)128 * 128 * 2);
  _Float16* bw1T = (_Float16*)carve((size_t)128 * 512 * 2);
  _Float16* bw2T = (_Float16*)carve((size_t)512 * 128 * 2);
  _Float16* gwqT = (_Float16*)carve((size_t)128 * 128 * 2);
  _Float16* gwkT = (_Float16*)carve((size_t)128 * 32 * 2);
  _Float16* gwvT = (_Float16*)carve((size_t)128 * 32 * 2);
  _Float16* gwoT = (_Float16*)carve((size_t)128 * 128 * 2);
  _Float16* gw1T = (_Float16*)carve((size_t)128 * 512 * 2);
  _Float16* gw2T = (_Float16*)carve((size_t)512 * 128 * 2);
  float* t_blk = (float*)carve((size_t)4096 * 64 * 128 * 4);
  float* img   = (float*)carve((size_t)4 * 256 * 256 * 128 * 4);
  float* rtmp  = (float*)carve((size_t)4096 * 4 * 128 * 4);
  float* ravg  = (float*)carve((size_t)4 * 4 * 128 * 4);
  float* outimg = t_blk;  // t_blk dead after block stage -> reuse

  auto g = [](int n) { return (n + TPB - 1) / TPB; };
  cvt_f16<<<g(512 * 128), TPB, 0, stream>>>(pw1_w, pw1h, 512, 128, 0);
  cvt_f16<<<g(128 * 512), TPB, 0, stream>>>(pw2_w, pw2h, 128, 512, 0);
  cvt_f16<<<g(128 * 128), TPB, 0, stream>>>(b_wq, bwqT, 128, 128, 1);
  cvt_f16<<<g(128 * 32),  TPB, 0, stream>>>(b_wk, bwkT, 128, 32, 1);
  cvt_f16<<<g(128 * 32),  TPB, 0, stream>>>(b_wv, bwvT, 128, 32, 1);
  cvt_f16<<<g(128 * 128), TPB, 0, stream>>>(b_wo, bwoT, 128, 128, 1);
  cvt_f16<<<g(128 * 512), TPB, 0, stream>>>(bff_w1, bw1T, 128, 512, 1);
  cvt_f16<<<g(512 * 128), TPB, 0, stream>>>(bff_w2, bw2T, 512, 128, 1);
  cvt_f16<<<g(128 * 128), TPB, 0, stream>>>(g_wq, gwqT, 128, 128, 1);
  cvt_f16<<<g(128 * 32),  TPB, 0, stream>>>(g_wk, gwkT, 128, 32, 1);
  cvt_f16<<<g(128 * 32),  TPB, 0, stream>>>(g_wv, gwvT, 128, 32, 1);
  cvt_f16<<<g(128 * 128), TPB, 0, stream>>>(g_wo, gwoT, 128, 128, 1);
  cvt_f16<<<g(128 * 512), TPB, 0, stream>>>(gff_w1, gw1T, 128, 512, 1);
  cvt_f16<<<g(512 * 128), TPB, 0, stream>>>(gff_w2, gw2T, 512, 128, 1);

  const int convShm = 267008;  // fits 320KB WGP LDS
  const int xfShm   = 183296;
  (void)hipFuncSetAttribute((const void*)conv_fused,
                            hipFuncAttributeMaxDynamicSharedMemorySize, convShm);
  (void)hipFuncSetAttribute((const void*)xformer<0>,
                            hipFuncAttributeMaxDynamicSharedMemorySize, xfShm);
  (void)hipFuncSetAttribute((const void*)xformer<1>,
                            hipFuncAttributeMaxDynamicSharedMemorySize, xfShm);

  conv_fused<<<4096, TPB, convShm, stream>>>(x, pw1h, pw1_g, pw1_b,
                                             dw_w, dw_g, dw_b,
                                             pw2h, pw2_g, pw2_b, t_blk);
  xformer<0><<<4096, TPB, xfShm, stream>>>(t_blk, reg,
                                           bwqT, bwkT, bwvT, bwoT,
                                           bln_g, bln_b, bff_lg, bff_lb,
                                           bw1T, bff_b1, bw2T, bff_b2,
                                           img, rtmp);
  reg_avg<<<g(4 * 4 * 128), TPB, 0, stream>>>(rtmp, ravg);
  xformer<1><<<4096, TPB, xfShm, stream>>>(img, ravg,
                                           gwqT, gwkT, gwvT, gwoT,
                                           gln_g, gln_b, gff_lg, gff_lb,
                                           gw1T, gff_b1, gw2T, gff_b2,
                                           outimg, nullptr);
  to_nchw<<<4 * 256 * 8, TPB, 0, stream>>>(outimg, (float*)d_out);
}
